// DNC_75342316306825
// MI455X (gfx1250) — compile-verified, matrix operand order checked by reading, not run on hardware
//
#include <hip/hip_runtime.h>
#include <cmath>

// ---------------- problem constants ----------------
// B=4096, N=128, W=64, R=4, H=512, IN=256, OUT=256, IFACE=471 (pad 480)
// iface part offsets: read_keys[0,256) read_str[256,260) write_key[260,324)
// write_str[324] erase[325,389) write_vec[389,453) free_g[453,457)
// alloc_g[457] write_g[458] read_modes[459,471)

typedef _Float16 f16;
typedef _Float16 v16h __attribute__((ext_vector_type(16)));
typedef float    v8f  __attribute__((ext_vector_type(8)));
typedef int      v4i  __attribute__((ext_vector_type(4)));

#define DEV __device__ __forceinline__

// ---------------- CDNA5 async global->LDS copy ----------------
#if defined(__has_builtin)
#  if __has_builtin(__builtin_amdgcn_global_load_async_to_lds_b128)
#    define HAVE_ASYNC_LDS 1
#  endif
#endif
#ifndef HAVE_ASYNC_LDS
#  define HAVE_ASYNC_LDS 0
#endif

typedef __attribute__((address_space(1))) v4i* gvec4p;
typedef __attribute__((address_space(3))) v4i* lvec4p;

DEV void copy16(const void* g, void* l){
#if HAVE_ASYNC_LDS
  void* gg = const_cast<void*>(g);
  __builtin_amdgcn_global_load_async_to_lds_b128((gvec4p)gg, (lvec4p)l, 0, 0);
#else
  *(v4i*)l = *(const v4i*)g;
#endif
}
DEV void async_wait(){
#if HAVE_ASYNC_LDS
#  if __has_builtin(__builtin_amdgcn_s_wait_asynccnt)
  __builtin_amdgcn_s_wait_asynccnt(0);
#  else
  asm volatile("s_wait_asynccnt 0" ::: "memory");
#  endif
#endif
}

DEV float sigm(float x){ return 1.0f/(1.0f+expf(-x)); }
DEV float oneplus(float x){ return (x>20.f) ? (1.f+x) : (1.f + log1pf(expf(x))); }

DEV v8f wmma_f16(v16h a, v16h b, v8f c){
  return __builtin_amdgcn_wmma_f32_16x16x32_f16(false, a, false, b, (short)0, c, false, false);
}

// 16-bit A/B fragment, operand stored row-major (n, k), contiguous K.
// per-lane row n; K = {k0+half*8 .. +7, k0+16+half*8 .. +7}  (ISA layout)
DEV v16h frag_nk(const f16* base, int ld, int n, int k0, int half){
  v16h v;
  const f16* p = base + (size_t)n*ld + (k0 + half*8);
#pragma unroll
  for (int j=0;j<8;++j) v[j]   = p[j];
#pragma unroll
  for (int j=0;j<8;++j) v[8+j] = p[16+j];
  return v;
}

// ---------------- prep: f32 -> f16 packing ----------------
__global__ void k_prep_xcat(const float* __restrict__ x, const float* __restrict__ rd,
                            const float* __restrict__ h, f16* __restrict__ xcat){
  int idx = blockIdx.x*256 + threadIdx.x;          // B*1024
  int b = idx>>10, k = idx&1023;
  float v = (k<256) ? x[b*256+k] : (k<512) ? rd[b*256+(k-256)] : h[b*512+(k-512)];
  xcat[idx] = (f16)v;
}
__global__ void k_prep_wcat(const float* __restrict__ W_ih, const float* __restrict__ W_hh,
                            f16* __restrict__ wcat){
  int idx = blockIdx.x*256 + threadIdx.x;          // 2048*1024
  int n = idx>>10, k = idx&1023;
  float v = (k<512) ? W_ih[n*512+k] : W_hh[n*512+(k-512)];
  wcat[idx] = (f16)v;
}
// W_iface (512 k, 471 n) -> transposed padded (480 n, 512 k)
__global__ void k_prep_wifT(const float* __restrict__ W_if, f16* __restrict__ wifT){
  int idx = blockIdx.x*256 + threadIdx.x;          // 480*512
  int n = idx>>9, k = idx&511;
  wifT[idx] = (f16)((n<471) ? W_if[k*471+n] : 0.f);
}
__global__ void k_prep_wlin(const float* __restrict__ W_lin, f16* __restrict__ wlin){
  int idx = blockIdx.x*256 + threadIdx.x;          // 256*768
  wlin[idx] = (f16)W_lin[idx];
}

// ---------------- fused LSTM GEMM + activation ----------------
// gates(4096x2048) = xcat(4096x1024) @ wcatT
// Block: 32-row A tile (64KB LDS) shared by 8 waves; each wave: 2 M-tiles x
// 4 gate-column tiles, ping-pong B buffers, 8 WMMA per 32-K step.
__global__ void __launch_bounds__(256) k_lstm_gemm(
    const f16* __restrict__ xcat, const f16* __restrict__ wcat,
    const float* __restrict__ b_ih, const float* __restrict__ b_hh,
    const float* __restrict__ c_old,
    float* __restrict__ h_new, float* __restrict__ c_new, f16* __restrict__ hr_h){
  __shared__ __align__(16) f16 atile[32*1024];     // 64 KB
  int tid  = threadIdx.x;
  int warp = tid>>5, lane = tid&31;
  int half = lane>>4, r = lane&15;
  int mpair = blockIdx.x>>2;                       // 128 m-pairs (32 rows each)
  int jt    = (blockIdx.x&3)*8 + warp;             // 32 column tiles
  int m0 = mpair*32, j0 = jt*16;
  { // stage A tile (contiguous 64KB) via async copy
    const char* g = (const char*)(xcat + (size_t)m0*1024);
    char* l = (char*)atile;
    for (int off = tid*16; off < 65536; off += 4096) copy16(g+off, l+off);
    async_wait();
  }
  __syncthreads();

  v8f acc[2][4];
#pragma unroll
  for (int t=0; t<2; ++t)
#pragma unroll
    for (int g=0; g<4; ++g)
#pragma unroll
      for (int e=0; e<8; ++e) acc[t][g][e] = 0.f;

  v16h b0[4], b1[4];
#pragma unroll
  for (int g=0; g<4; ++g) b0[g] = frag_nk(wcat, 1024, j0 + r + g*512, 0, half);
  for (int k0=0; k0<1024; k0+=64){
#pragma unroll
    for (int g=0; g<4; ++g) b1[g] = frag_nk(wcat, 1024, j0 + r + g*512, k0+32, half);
    {
      v16h a0 = frag_nk(atile, 1024, r,      k0, half);
      v16h a1 = frag_nk(atile, 1024, 16 + r, k0, half);
#pragma unroll
      for (int g=0; g<4; ++g){
        acc[0][g] = wmma_f16(a0, b0[g], acc[0][g]);
        acc[1][g] = wmma_f16(a1, b0[g], acc[1][g]);
      }
    }
    if (k0 + 64 < 1024){
#pragma unroll
      for (int g=0; g<4; ++g) b0[g] = frag_nk(wcat, 1024, j0 + r + g*512, k0+64, half);
    }
    {
      v16h a0 = frag_nk(atile, 1024, r,      k0+32, half);
      v16h a1 = frag_nk(atile, 1024, 16 + r, k0+32, half);
#pragma unroll
      for (int g=0; g<4; ++g){
        acc[0][g] = wmma_f16(a0, b1[g], acc[0][g]);
        acc[1][g] = wmma_f16(a1, b1[g], acc[1][g]);
      }
    }
  }

  int col = j0 + r;
  float bi = b_ih[col]      + b_hh[col];
  float bf = b_ih[col+512]  + b_hh[col+512];
  float bg = b_ih[col+1024] + b_hh[col+1024];
  float bo = b_ih[col+1536] + b_hh[col+1536];
#pragma unroll
  for (int t=0; t<2; ++t){
#pragma unroll
    for (int v=0; v<8; ++v){
      int row = m0 + t*16 + half*8 + v;
      float iv = acc[t][0][v]+bi, fv = acc[t][1][v]+bf;
      float gv = acc[t][2][v]+bg, ov = acc[t][3][v]+bo;
      float cn = sigm(fv)*c_old[(size_t)row*512+col] + sigm(iv)*tanhf(gv);
      float hn = sigm(ov)*tanhf(cn);
      c_new[(size_t)row*512+col] = cn;
      h_new[(size_t)row*512+col] = hn;
      hr_h[(size_t)row*768+col]  = (f16)hn;
    }
  }
}

// ---------------- iface GEMM: h_new(4096x512) @ wifT(480x512)T ----------------
__global__ void __launch_bounds__(256) k_iface_gemm(
    const f16* __restrict__ hr, const f16* __restrict__ wifT,
    const float* __restrict__ b_if, float* __restrict__ iface){
  __shared__ __align__(16) f16 atile[16*512];      // 16 KB
  int tid  = threadIdx.x;
  int lane = tid & 31;
  int half = lane>>4, r = lane&15;
  int m0 = blockIdx.x*16;                          // 256 row tiles
  int nt = blockIdx.y*8 + (tid>>5);                // 30 of 32 col tiles active
  { // stage A tile rows (stride 768 in hr_h -> copy row by row)
    for (int c = tid; c < 1024; c += 256){         // 1024 16B chunks (64/row)
      int row = c>>6, off = (c&63)*16;
      copy16((const char*)(hr + (size_t)(m0+row)*768) + off,
             (char*)atile + row*1024 + off);
    }
    async_wait();
  }
  __syncthreads();
  if (nt >= 30) return;                            // whole-wave exit
  int n0 = nt*16;

  v8f acc;
#pragma unroll
  for (int e=0; e<8; ++e) acc[e] = 0.f;
  v16h b0 = frag_nk(wifT, 512, n0+r, 0, half), b1;
  for (int k0=0; k0<512; k0+=64){
    b1 = frag_nk(wifT, 512, n0+r, k0+32, half);
    { v16h a = frag_nk(atile, 512, r, k0, half);    acc = wmma_f16(a, b0, acc); }
    if (k0 + 64 < 512) b0 = frag_nk(wifT, 512, n0+r, k0+64, half);
    { v16h a = frag_nk(atile, 512, r, k0+32, half); acc = wmma_f16(a, b1, acc); }
  }
  int col = n0 + r;
  float bias = (col<471) ? b_if[col] : 0.f;
#pragma unroll
  for (int v=0; v<8; ++v){
    int row = m0 + half*8 + v;
    iface[(size_t)row*480 + col] = acc[v] + bias;
  }
}

// ---------------- write weights / usage / allocation / precedence ----------------
__global__ void __launch_bounds__(128) k_write_weights(
    const float* __restrict__ mem, const float* __restrict__ iface,
    const float* __restrict__ r_weights, const float* __restrict__ w_weights,
    const float* __restrict__ usage, const float* __restrict__ precedence,
    float* __restrict__ u_out, float* __restrict__ ww_out, float* __restrict__ prec_out){
  int b = blockIdx.x, n = threadIdx.x;             // 128 threads
  const float* ifr = iface + (size_t)b*480;
  __shared__ float red[128];
  __shared__ float shv[128];
  __shared__ float srt[128];
  __shared__ float cp[128];

  float kk = 0.f;
  for (int w=0; w<64; ++w){ float kv = ifr[260+w]; kk += kv*kv; }
  float kinv = 1.f/(sqrtf(kk)+1e-6f);
  const float* mrow = mem + ((size_t)b*128 + n)*64;
  float mm = 0.f, dot = 0.f;
  for (int w=0; w<64; ++w){ float mv = mrow[w]; mm += mv*mv; dot += mv*ifr[260+w]; }
  float logit = oneplus(ifr[324]) * dot * kinv / (sqrtf(mm)+1e-6f);
  red[n] = logit; __syncthreads();
  for (int s=64; s>0; s>>=1){ if (n<s) red[n] = fmaxf(red[n], red[n+s]); __syncthreads(); }
  float mx = red[0]; __syncthreads();
  float ex = expf(logit - mx);
  red[n] = ex; __syncthreads();
  for (int s=64; s>0; s>>=1){ if (n<s) red[n] += red[n+s]; __syncthreads(); }
  float cw = ex / red[0]; __syncthreads();

  float psi = 1.f;
#pragma unroll
  for (int r=0; r<4; ++r)
    psi *= 1.f - sigm(ifr[453+r]) * r_weights[((size_t)b*4 + r)*128 + n];
  float us = usage[(size_t)b*128+n], wo = w_weights[(size_t)b*128+n];
  float uu = (us + wo - us*wo) * psi;
  u_out[(size_t)b*128+n] = uu;

  shv[n] = uu; __syncthreads();
  int rank = 0;
  for (int j=0; j<128; ++j){
    float uj = shv[j];
    rank += (uj < uu) || (uj == uu && j < n);
  }
  srt[rank] = uu; __syncthreads();
  if (n==0){ float p = 1.f; for (int j=0; j<128; ++j){ cp[j] = p; p *= srt[j]; } }
  __syncthreads();
  float alloc = (1.f - uu) * cp[rank];

  float ag = sigm(ifr[457]), wg = sigm(ifr[458]);
  float wwv = wg * (ag*alloc + (1.f-ag)*cw);
  ww_out[(size_t)b*128+n] = wwv;

  red[n] = wwv; __syncthreads();
  for (int s=64; s>0; s>>=1){ if (n<s) red[n] += red[n+s]; __syncthreads(); }
  prec_out[(size_t)b*128+n] = (1.f - red[0]) * precedence[(size_t)b*128+n] + wwv;
}

// ---------------- memory update ----------------
__global__ void k_mem(const float* __restrict__ mem, const float* __restrict__ iface,
                      const float* __restrict__ ww, float* __restrict__ mem_new){
  size_t idx = (size_t)blockIdx.x*256 + threadIdx.x;   // B*N*W
  int w = idx & 63; int n = (int)(idx>>6) & 127; int b = (int)(idx>>13);
  const float* ifr = iface + (size_t)b*480;
  float e  = sigm(ifr[325+w]);
  float vv = ifr[389+w];
  float wn = ww[(size_t)b*128+n];
  mem_new[idx] = mem[idx]*(1.f - wn*e) + wn*vv;
}

// ---------------- link matrix update ----------------
__global__ void k_link(const float* __restrict__ link, const float* __restrict__ ww,
                       const float* __restrict__ prec, float* __restrict__ L){
  size_t idx = (size_t)blockIdx.x*256 + threadIdx.x;   // B*N*N
  int j = idx & 127; int i = (int)(idx>>7) & 127; int b = (int)(idx>>14);
  float wi = ww[(size_t)b*128+i], wj = ww[(size_t)b*128+j];
  float res = (i==j) ? 0.f : (1.f - wi - wj)*link[idx] + wi*prec[(size_t)b*128+j];
  L[idx] = res;
}

// ---------------- read content weights on mem_new ----------------
__global__ void __launch_bounds__(128) k_read_cw(
    const float* __restrict__ mem_new, const float* __restrict__ iface,
    float* __restrict__ cr){
  int b = blockIdx.x, n = threadIdx.x;
  const float* ifr = iface + (size_t)b*480;
  const float* mrow = mem_new + ((size_t)b*128 + n)*64;
  __shared__ float red[128];
  float mm = 0.f;
  for (int w=0; w<64; ++w){ float mv = mrow[w]; mm += mv*mv; }
  float minv = 1.f/(sqrtf(mm)+1e-6f);
  for (int r=0; r<4; ++r){
    const float* key = ifr + r*64;
    float kk = 0.f, dot = 0.f;
    for (int w=0; w<64; ++w){ float kv = key[w]; kk += kv*kv; dot += kv*mrow[w]; }
    float logit = oneplus(ifr[256+r]) * dot * minv / (sqrtf(kk)+1e-6f);
    red[n] = logit; __syncthreads();
    for (int s=64; s>0; s>>=1){ if (n<s) red[n] = fmaxf(red[n], red[n+s]); __syncthreads(); }
    float mx = red[0]; __syncthreads();
    float ex = expf(logit - mx);
    red[n] = ex; __syncthreads();
    for (int s=64; s>0; s>>=1){ if (n<s) red[n] += red[n+s]; __syncthreads(); }
    cr[((size_t)b*4 + r)*128 + n] = ex / red[0];
    __syncthreads();
  }
}

// ---------------- fwd/bwd link einsums + read-mode mix (LDS-staged L) ----------------
__global__ void __launch_bounds__(128) k_rw(
    const float* __restrict__ L, const float* __restrict__ r_weights,
    const float* __restrict__ cr, const float* __restrict__ iface,
    float* __restrict__ rw_new){
  __shared__ __align__(16) float Lt[128*132];      // padded stride vs bank conflicts
  __shared__ float shrw[128];
  int b = blockIdx.x, n = threadIdx.x;             // 128 threads
  const char* g = (const char*)(L + (size_t)b*16384);
  for (int c = n; c < 4096; c += 128){             // 4096 x 16B chunks
    int row = c>>5, off = (c&31)*16;               // 32 chunks per 512B row
    copy16(g + (size_t)row*512 + off, (char*)Lt + (size_t)row*528 + off);
  }
  async_wait();
  __syncthreads();
  const float* ifr = iface + (size_t)b*480;
  for (int r=0; r<4; ++r){
    shrw[n] = r_weights[((size_t)b*4 + r)*128 + n];
    __syncthreads();
    float fw = 0.f, bw = 0.f;
    for (int m=0; m<128; ++m){
      float rv = shrw[m];
      fw += Lt[n*132 + m] * rv;                    // fwd: L[n,m]
      bw += Lt[m*132 + n] * rv;                    // bwd: L[m,n]
    }
    float a0 = ifr[459 + r*3 + 0], a1 = ifr[459 + r*3 + 1], a2 = ifr[459 + r*3 + 2];
    float mx = fmaxf(a0, fmaxf(a1, a2));
    float e0 = expf(a0-mx), e1 = expf(a1-mx), e2 = expf(a2-mx);
    float si = 1.f/(e0+e1+e2);
    float crv = cr[((size_t)b*4 + r)*128 + n];
    rw_new[((size_t)b*4 + r)*128 + n] = (e0*bw + e1*crv + e2*fw) * si;
    __syncthreads();
  }
}

// ---------------- read vectors (LDS-staged mem_new tile) ----------------
__global__ void __launch_bounds__(256) k_read_new(
    const float* __restrict__ rw_new, const float* __restrict__ mem_new,
    float* __restrict__ read_new, f16* __restrict__ hr_h){
  __shared__ __align__(16) float mtile[128*64];    // 32 KB
  __shared__ float shrw[512];
  int b = blockIdx.x, t = threadIdx.x;             // 256 threads
  const char* g = (const char*)(mem_new + (size_t)b*8192);
  for (int off = t*16; off < 32768; off += 4096) copy16(g+off, (char*)mtile+off);
  shrw[t]     = rw_new[(size_t)b*512 + t];
  shrw[t+256] = rw_new[(size_t)b*512 + 256 + t];
  async_wait();
  __syncthreads();
  int r = t>>6, w = t&63;
  float acc = 0.f;
  for (int n=0; n<128; ++n)
    acc += shrw[r*128 + n] * mtile[n*64 + w];
  read_new[(size_t)b*256 + t] = acc;
  hr_h[(size_t)b*768 + 512 + t] = (f16)acc;
}

// ---------------- final GEMM: [h|read](4096x768) @ W_linT(256x768) ----------------
__global__ void __launch_bounds__(256) k_out_gemm(
    const f16* __restrict__ hr, const f16* __restrict__ wlin,
    const float* __restrict__ b_lin, float* __restrict__ out){
  __shared__ __align__(16) f16 atile[16*768];      // 24 KB
  int tid  = threadIdx.x;
  int wave = blockIdx.x*8 + (tid>>5);
  int lane = tid & 31;
  int half = lane>>4, r = lane&15;
  int nt = wave&15;                                // 256 x 16 tiles; block shares mt
  int m0 = (blockIdx.x>>1)*16, n0 = nt*16;
  { // stage A tile (contiguous 24KB)
    const char* g = (const char*)(hr + (size_t)m0*768);
    char* l = (char*)atile;
    for (int off = tid*16; off < 24576; off += 4096) copy16(g+off, l+off);
    async_wait();
  }
  __syncthreads();

  v8f acc;
#pragma unroll
  for (int e=0; e<8; ++e) acc[e] = 0.f;
  v16h b0 = frag_nk(wlin, 768, n0+r, 0, half), b1;
  for (int k0=0; k0<768; k0+=64){
    b1 = frag_nk(wlin, 768, n0+r, k0+32, half);
    { v16h a = frag_nk(atile, 768, r, k0, half);    acc = wmma_f16(a, b0, acc); }
    if (k0 + 64 < 768) b0 = frag_nk(wlin, 768, n0+r, k0+64, half);
    { v16h a = frag_nk(atile, 768, r, k0+32, half); acc = wmma_f16(a, b1, acc); }
  }
  int col = n0 + r;
  float bias = b_lin[col];
#pragma unroll
  for (int v=0; v<8; ++v){
    int row = m0 + half*8 + v;
    out[(size_t)row*256 + col] = acc[v] + bias;
  }
}

// ---------------- launcher ----------------
extern "C" void kernel_launch(void* const* d_in, const int* in_sizes, int n_in,
                              void* d_out, int out_size, void* d_ws, size_t ws_size,
                              hipStream_t stream){
  (void)in_sizes; (void)n_in; (void)out_size; (void)ws_size;
  const float* x     = (const float*)d_in[0];
  const float* rd    = (const float*)d_in[1];
  const float* h     = (const float*)d_in[2];
  const float* c     = (const float*)d_in[3];
  const float* mem   = (const float*)d_in[4];
  const float* link  = (const float*)d_in[5];
  const float* rwts  = (const float*)d_in[6];
  const float* wwts  = (const float*)d_in[7];
  const float* usage = (const float*)d_in[8];
  const float* prec  = (const float*)d_in[9];
  const float* W_ih  = (const float*)d_in[10];
  const float* b_ih  = (const float*)d_in[11];
  const float* W_hh  = (const float*)d_in[12];
  const float* b_hh  = (const float*)d_in[13];
  const float* W_if  = (const float*)d_in[14];
  const float* b_if  = (const float*)d_in[15];
  const float* W_lin = (const float*)d_in[16];
  const float* b_lin = (const float*)d_in[17];

  float* O = (float*)d_out;          // outputs concatenated in return order
  float* out_p      = O;
  float* read_new_p = O + 1048576;
  float* h_new_p    = O + 2097152;
  float* c_new_p    = O + 4194304;
  float* mem_new_p  = O + 6291456;
  float* L_p        = O + 39845888;
  float* rw_new_p   = O + 106954752;
  float* ww_p       = O + 109051904;
  float* u_p        = O + 109576192;
  float* prec_p     = O + 110100480;

  char* ws = (char*)d_ws;            // ~36 MB scratch
  f16*   xcat_h = (f16*)(ws + 0);              // 4096*1024 f16
  f16*   wcat_h = (f16*)(ws + 8388608);        // 2048*1024 f16
  f16*   wifT_h = (f16*)(ws + 12582912);       // 480*512 f16 (transposed, padded)
  f16*   wlin_h = (f16*)(ws + 13074432);       // 256*768 f16
  f16*   hr_h   = (f16*)(ws + 13467648);       // 4096*768 f16
  float* iface  = (float*)(ws + 19759104);     // 4096*480 f32
  float* cr     = (float*)(ws + 27623424);     // 4096*4*128 f32

  k_prep_xcat<<<16384, 256, 0, stream>>>(x, rd, h, xcat_h);
  k_prep_wcat<<<8192, 256, 0, stream>>>(W_ih, W_hh, wcat_h);
  k_prep_wifT<<<960, 256, 0, stream>>>(W_if, wifT_h);
  k_prep_wlin<<<768, 256, 0, stream>>>(W_lin, wlin_h);

  k_lstm_gemm<<<512, 256, 0, stream>>>(xcat_h, wcat_h, b_ih, b_hh, c,
                                       h_new_p, c_new_p, hr_h);
  k_iface_gemm<<<dim3(256,4), 256, 0, stream>>>(hr_h, wifT_h, b_if, iface);

  k_write_weights<<<4096, 128, 0, stream>>>(mem, iface, rwts, wwts, usage, prec,
                                            u_p, ww_p, prec_p);
  k_mem <<<131072, 256, 0, stream>>>(mem, iface, ww_p, mem_new_p);
  k_link<<<262144, 256, 0, stream>>>(link, ww_p, prec, L_p);

  k_read_cw<<<4096, 128, 0, stream>>>(mem_new_p, iface, cr);
  k_rw     <<<4096, 128, 0, stream>>>(L_p, rwts, cr, iface, rw_new_p);
  k_read_new<<<4096, 256, 0, stream>>>(rw_new_p, mem_new_p, read_new_p, hr_h);

  k_out_gemm<<<512, 256, 0, stream>>>(hr_h, wlin_h, b_lin, out_p);
}
